// GNN_85942295593181
// MI455X (gfx1250) — matrix operand order repaired and checked by reading, not compiled, not run on hardware
//
#include <hip/hip_runtime.h>
#include <cstdint>

// ---------------------------------------------------------------------------
// GAT pipeline for MI455X (gfx1250, wave32).
// Dense GEMMs: bf16 WMMA (v_wmma_f32_16x16x32_bf16), fp32 accumulate.
//   - 16x64 output per wave (A-fragment reuse x4)
//   - B slab (Kx64, shared by all 8 waves of the block) staged into LDS by the
//     Tensor Data Mover (tensor_load_to_lds), double buffered, s_wait_tensorcnt
// Attention softmax-scatter: fp32 global atomics + int-key atomic max.
// fc1 node part factored out of the per-edge GEMM (broadcast row).
// ---------------------------------------------------------------------------

typedef __attribute__((ext_vector_type(16))) __bf16 v16bf;
typedef __attribute__((ext_vector_type(8)))  __bf16 v8bf;
typedef __attribute__((ext_vector_type(8)))  float  v8f;
typedef __attribute__((ext_vector_type(4)))  unsigned v4u;
typedef __attribute__((ext_vector_type(8)))  int      v8i;
typedef __attribute__((ext_vector_type(4)))  int      v4i;

#define GNN_N   16384
#define GNN_E   131072
#define GNN_EDG (GNN_E + GNN_N)   // edges + self loops
#define GNN_H   4

// Order-preserving map float -> uint32 for atomicMax-based segment max.
__device__ __forceinline__ unsigned gnn_fkey(float f) {
  unsigned b = __float_as_uint(f);
  return (b & 0x80000000u) ? ~b : (b | 0x80000000u);
}
__device__ __forceinline__ float gnn_funkey(unsigned k) {
  unsigned b = (k & 0x80000000u) ? (k ^ 0x80000000u) : ~k;
  return __uint_as_float(b);
}

// ---------------------------------------------------------------------------
// TDM: issue a 2D tensor_load_to_lds of a 64-row x 32-element bf16 tile.
//   gsrc   : global address of tile origin (row-major, row stride = K elems)
//   ldsoff : destination LDS byte offset
// D# layout per CDNA5 ISA ch.8 (group0: ctrl/lds/global/type, group1: dims).
// ---------------------------------------------------------------------------
__device__ __forceinline__ void gnn_tdm_load_b_tile(const __bf16* gsrc,
                                                    unsigned ldsoff,
                                                    int K, int M) {
  uint64_t ga = (uint64_t)(uintptr_t)gsrc;
  v4u g0 = { 1u,                                   // count=1 (valid), user mode
             ldsoff,                               // lds_addr [63:32]
             (unsigned)ga,                         // global_addr lo
             (unsigned)(ga >> 32) | (2u << 30) };  // global_addr hi | type=2
  unsigned td0 = (unsigned)K;    // tensor_dim0 (elements)
  unsigned td1 = (unsigned)M;    // tensor_dim1 (rows)
  uint64_t ts0 = (uint64_t)K;    // tensor_dim0_stride (elements)
  v8i g1 = { (int)(1u << 16),                                  // data_size=1 (2B)
             (int)((td0 & 0xFFFFu) << 16),                     // td0[15:0] @ [63:48]
             (int)((td0 >> 16) | ((td1 & 0xFFFFu) << 16)),     // td0[31:16], td1[15:0]
             (int)((td1 >> 16) | (32u << 16)),                 // td1[31:16], tile_dim0=32
             (int)64,                                          // tile_dim1=64, tile_dim2=0
             (int)(unsigned)ts0,                               // ts0[31:0]
             (int)((unsigned)(ts0 >> 32) & 0xFFFFu),           // ts0[47:32], ts1=0
             0 };
  v4i g2 = { 0, 0, 0, 0 };
  v4i g3 = { 0, 0, 0, 0 };
#if __clang_major__ >= 23
  v8i g4 = { 0, 0, 0, 0, 0, 0, 0, 0 };
  __builtin_amdgcn_tensor_load_to_lds(g0, g1, g2, g3, g4, 0);
#else
  __builtin_amdgcn_tensor_load_to_lds(g0, g1, g2, g3, 0);
#endif
}

// ---------------------------------------------------------------------------
// Generic helpers
// ---------------------------------------------------------------------------
__global__ void gnn_zero_u32(unsigned* __restrict__ p, size_t n) {
  size_t i = (size_t)blockIdx.x * blockDim.x + threadIdx.x;
  size_t stride = (size_t)gridDim.x * blockDim.x;
  for (; i < n; i += stride) p[i] = 0u;
}

__global__ void gnn_f32_to_bf16(const float* __restrict__ s, __bf16* __restrict__ d, size_t n) {
  size_t i = (size_t)blockIdx.x * blockDim.x + threadIdx.x;
  if (i < n) d[i] = (__bf16)s[i];
}

// ---------------------------------------------------------------------------
// WMMA GEMM:  C[nrows,M] = A[nrows,K] * B[M,K]^T   (bf16 in, bf16 out, f32 acc)
// Block = 256 threads (8 waves) -> 128 rows x 64 cols of C.
//   wave w: rows [rb*128 + w*16, +16), cols [cg*64, +64)  (4 accumulator tiles)
// B slab cg (Kx64) is DMA'd 32-K-columns at a time into LDS (double buffer)
// by wave 0 via the Tensor Data Mover; consumed by all 8 waves via ds loads.
// A fragment (16x32): lanes 0-15 row=lane, K=0..15; lanes 16-31 K=16..31
// B fragment (32x16): col=lane&15; lanes 0-15 K=0..15, lanes 16-31 K=16..31
// ---------------------------------------------------------------------------
__global__ void gnn_gemm_bf16nt(const __bf16* __restrict__ A,
                                const __bf16* __restrict__ B,
                                __bf16* __restrict__ C,
                                int nrows, int K, int M) {
  __shared__ __bf16 bs[2][64][32];   // 8 KB double buffer: [buf][row(=C col)][k]
  int colGroups = M >> 6;
  int rb = blockIdx.x / colGroups;
  int cg = blockIdx.x - rb * colGroups;
  int wave = threadIdx.x >> 5;
  int lane = threadIdx.x & 31;
  int lr = lane & 15;
  int hi = lane >> 4;                 // 0: lanes 0-15, 1: lanes 16-31
  int row0 = (rb << 7) + (wave << 4);
  int col0 = cg << 6;
  const __bf16* arow  = A + (size_t)(row0 + lr) * K + (hi ? 16 : 0);
  const __bf16* btile = B + (size_t)col0 * K;   // slab origin (row stride K)
  unsigned lds0 = (unsigned)(uintptr_t)(&bs[0][0][0]);
  unsigned lds1 = (unsigned)(uintptr_t)(&bs[1][0][0]);
  int numIt = K >> 5;
  bool issuer = (threadIdx.x < 32);   // wave 0 drives the TDM

  if (issuer) {
    gnn_tdm_load_b_tile(btile, lds0, K, M);
    if (numIt > 1) gnn_tdm_load_b_tile(btile + 32, lds1, K, M);
  }

  v8f acc0 = {}, acc1 = {}, acc2 = {}, acc3 = {};
  for (int it = 0; it < numIt; ++it) {
    if (issuer) {
      if (it + 2 <= numIt) __builtin_amdgcn_s_wait_tensorcnt(1);
      else                 __builtin_amdgcn_s_wait_tensorcnt(0);
    }
    __syncthreads();                  // buffer (it&1) is ready for everyone

    int k0 = it << 5;
    v8bf a0 = *(const v8bf*)(arow + k0);
    v8bf a1 = *(const v8bf*)(arow + k0 + 8);
    v16bf af;
#pragma unroll
    for (int i = 0; i < 8; ++i) { af[i] = a0[i]; af[i + 8] = a1[i]; }

    const __bf16* bb = &bs[it & 1][0][0];
    v16bf b0 = *(const v16bf*)(bb + (0 * 16 + lr) * 32 + (hi ? 16 : 0));
    v16bf b1 = *(const v16bf*)(bb + (1 * 16 + lr) * 32 + (hi ? 16 : 0));
    v16bf b2 = *(const v16bf*)(bb + (2 * 16 + lr) * 32 + (hi ? 16 : 0));
    v16bf b3 = *(const v16bf*)(bb + (3 * 16 + lr) * 32 + (hi ? 16 : 0));
    acc0 = __builtin_amdgcn_wmma_f32_16x16x32_bf16(false, af, false, b0, (short)0, acc0, false, false);
    acc1 = __builtin_amdgcn_wmma_f32_16x16x32_bf16(false, af, false, b1, (short)0, acc1, false, false);
    acc2 = __builtin_amdgcn_wmma_f32_16x16x32_bf16(false, af, false, b2, (short)0, acc2, false, false);
    acc3 = __builtin_amdgcn_wmma_f32_16x16x32_bf16(false, af, false, b3, (short)0, acc3, false, false);

    __syncthreads();                  // everyone done reading buffer (it&1)
    if (issuer && (it + 2) < numIt)
      gnn_tdm_load_b_tile(btile + ((it + 2) << 5), (it & 1) ? lds1 : lds0, K, M);
  }

#pragma unroll
  for (int r = 0; r < 8; ++r) {
    size_t row = (size_t)(row0 + r + (hi << 3));
    __bf16* crow = C + row * M + col0 + lr;
    crow[0]  = (__bf16)acc0[r];
    crow[16] = (__bf16)acc1[r];
    crow[32] = (__bf16)acc2[r];
    crow[48] = (__bf16)acc3[r];
  }
}

// ---------------------------------------------------------------------------
// Layer 1 "GEMM": K==1, so h[n, j] = x[n] * W1[j]
// ---------------------------------------------------------------------------
__global__ void gnn_l1(const float* __restrict__ x, const float* __restrict__ W1,
                       __bf16* __restrict__ h) {
  size_t i = (size_t)blockIdx.x * blockDim.x + threadIdx.x;
  if (i < (size_t)GNN_N * 512)
    h[i] = (__bf16)(x[i >> 9] * W1[i & 511]);
}

// ---------------------------------------------------------------------------
// Per-node attention scores: s_src[n,h]=sum_c h[n,h,c]*a_src[h,c] (and a_dst)
// Block = 128 threads = 4 waves; wave w handles head w.
// ---------------------------------------------------------------------------
__global__ void gnn_scores(const __bf16* __restrict__ h,
                           const float* __restrict__ asrc,
                           const float* __restrict__ adst,
                           float* __restrict__ ss, float* __restrict__ sd,
                           int C, int M) {
  int n = blockIdx.x;
  int head = threadIdx.x >> 5;
  int lane = threadIdx.x & 31;
  const __bf16* hp = h + (size_t)n * M + head * C;
  const float* as = asrc + head * C;
  const float* ad = adst + head * C;
  float s1 = 0.f, s2 = 0.f;
  for (int c = lane; c < C; c += 32) {
    float hv = (float)hp[c];
    s1 += hv * as[c];
    s2 += hv * ad[c];
  }
#pragma unroll
  for (int off = 16; off; off >>= 1) {
    s1 += __shfl_down(s1, off, 32);
    s2 += __shfl_down(s2, off, 32);
  }
  if (lane == 0) { ss[n * GNN_H + head] = s1; sd[n * GNN_H + head] = s2; }
}

// ---------------------------------------------------------------------------
// Edge pass A: e = leaky_relu(s_src[src]+s_dst[dst]); running max per (dst,head)
// ---------------------------------------------------------------------------
__global__ void gnn_edge_logits(const int* __restrict__ ei,
                                const float* __restrict__ ss,
                                const float* __restrict__ sd,
                                float* __restrict__ ebuf,
                                unsigned* __restrict__ mkey) {
  int i = blockIdx.x * blockDim.x + threadIdx.x;
  if (i >= GNN_EDG * GNN_H) return;
  int e = i >> 2, hd = i & 3;
  int s = (e < GNN_E) ? ei[e] : (e - GNN_E);
  int d = (e < GNN_E) ? ei[GNN_E + e] : (e - GNN_E);
  float v = ss[s * GNN_H + hd] + sd[d * GNN_H + hd];
  v = (v > 0.f) ? v : 0.2f * v;
  ebuf[i] = v;
  atomicMax(&mkey[d * GNN_H + hd], gnn_fkey(v));
}

// Edge pass B: ex = exp(e - m[dst]); denom[dst] += ex
__global__ void gnn_edge_exp(const int* __restrict__ ei,
                             float* __restrict__ ebuf,
                             const unsigned* __restrict__ mkey,
                             float* __restrict__ denom) {
  int i = blockIdx.x * blockDim.x + threadIdx.x;
  if (i >= GNN_EDG * GNN_H) return;
  int e = i >> 2, hd = i & 3;
  int d = (e < GNN_E) ? ei[GNN_E + e] : (e - GNN_E);
  float m = gnn_funkey(mkey[d * GNN_H + hd]);
  float ex = __expf(ebuf[i] - m);
  ebuf[i] = ex;
  atomicAdd(&denom[d * GNN_H + hd], ex);
}

// Edge pass C: acc[dst] += h[src] * alpha   (one block per edge)
__global__ void gnn_edge_scatter(const int* __restrict__ ei,
                                 const __bf16* __restrict__ hmat,
                                 const float* __restrict__ ebuf,
                                 const float* __restrict__ denom,
                                 float* __restrict__ acc,
                                 int M, int cshift) {
  int e = blockIdx.x;
  int s = (e < GNN_E) ? ei[e] : (e - GNN_E);
  int d = (e < GNN_E) ? ei[GNN_E + e] : (e - GNN_E);
  float al[GNN_H];
#pragma unroll
  for (int h2 = 0; h2 < GNN_H; ++h2)
    al[h2] = ebuf[e * GNN_H + h2] / denom[d * GNN_H + h2];
  for (int c = threadIdx.x; c < M; c += blockDim.x) {
    float hv = (float)hmat[(size_t)s * M + c];
    atomicAdd(&acc[(size_t)d * M + c], hv * al[c >> cshift]);
  }
}

// Finalize: v = elu(acc + bias); write bf16 act; optionally accumulate channel sums
__global__ void gnn_finalize(const float* __restrict__ acc,
                             const float* __restrict__ bias,
                             __bf16* __restrict__ act,
                             float* __restrict__ aggp,   // nullptr unless last layer
                             int mmask, size_t total) {
  size_t i = (size_t)blockIdx.x * blockDim.x + threadIdx.x;
  if (i >= total) return;
  int j = (int)(i & (size_t)mmask);
  float v = acc[i] + bias[j];
  v = (v > 0.f) ? v : expm1f(v);
  act[i] = (__bf16)v;
  if (aggp) atomicAdd(&aggp[j], v);
}

// z_node[o] = (agg/N) . fc1_w[o, :1024]   (node part of fc1, computed once)
__global__ void gnn_znode(const float* __restrict__ agg,
                          const float* __restrict__ fc1w,
                          float* __restrict__ znode) {
  int o = threadIdx.x;  // 256 threads
  const float* wrow = fc1w + (size_t)o * 1028;
  float ssum = 0.f;
  for (int k = 0; k < 1024; ++k) ssum += agg[k] * wrow[k];
  znode[o] = ssum * (1.0f / (float)GNN_N);
}

// z1[e,o] = relu(z_node[o] + edge_attr[e,:4].fc1_w[o,1024:] + fc1_b[o])  (bf16 out)
__global__ void gnn_fc1(const float* __restrict__ eattr,
                        const float* __restrict__ fc1w,
                        const float* __restrict__ fc1b,
                        const float* __restrict__ znode,
                        __bf16* __restrict__ z1) {
  int e = blockIdx.x;
  int o = threadIdx.x;  // 256
  const float* wrow = fc1w + (size_t)o * 1028 + 1024;
  float v = znode[o] + fc1b[o];
#pragma unroll
  for (int k = 0; k < 4; ++k) v += eattr[e * 4 + k] * wrow[k];
  z1[(size_t)e * 256 + o] = (__bf16)fmaxf(v, 0.f);
}

// out[e] = relu(z2[e,:]+fc2_b) . fc3_w + fc3_b   (block of 128 per edge)
__global__ void gnn_fc3(const __bf16* __restrict__ z2,
                        const float* __restrict__ fc2b,
                        const float* __restrict__ fc3w,
                        const float* __restrict__ fc3b,
                        float* __restrict__ out) {
  __shared__ float red[4];
  int e = blockIdx.x, t = threadIdx.x;  // 128 threads
  float v = (float)z2[(size_t)e * 128 + t] + fc2b[t];
  v = fmaxf(v, 0.f) * fc3w[t];
#pragma unroll
  for (int off = 16; off; off >>= 1) v += __shfl_down(v, off, 32);
  if ((t & 31) == 0) red[t >> 5] = v;
  __syncthreads();
  if (t == 0) out[e] = red[0] + red[1] + red[2] + red[3] + fc3b[0];
}

// ---------------------------------------------------------------------------
// Host launcher
// ---------------------------------------------------------------------------
extern "C" void kernel_launch(void* const* d_in, const int* in_sizes, int n_in,
                              void* d_out, int out_size, void* d_ws, size_t ws_size,
                              hipStream_t stream) {
  (void)in_sizes; (void)n_in; (void)out_size; (void)ws_size;
  const float* x     = (const float*)d_in[0];
  const int*   ei    = (const int*)d_in[1];
  const float* eattr = (const float*)d_in[2];
  const float* W1 = (const float*)d_in[3];
  const float* as1 = (const float*)d_in[4], *ad1 = (const float*)d_in[5], *b1 = (const float*)d_in[6];
  const float* W2 = (const float*)d_in[7];
  const float* as2 = (const float*)d_in[8], *ad2 = (const float*)d_in[9], *b2 = (const float*)d_in[10];
  const float* W3 = (const float*)d_in[11];
  const float* as3 = (const float*)d_in[12], *ad3 = (const float*)d_in[13], *b3 = (const float*)d_in[14];
  const float* W4 = (const float*)d_in[15];
  const float* as4 = (const float*)d_in[16], *ad4 = (const float*)d_in[17], *b4 = (const float*)d_in[18];
  const float* fc1w = (const float*)d_in[19], *fc1b = (const float*)d_in[20];
  const float* fc2w = (const float*)d_in[21], *fc2b = (const float*)d_in[22];
  const float* fc3w = (const float*)d_in[23], *fc3b = (const float*)d_in[24];

  // Workspace carve-out (~269 MB total)
  char* base = (char*)d_ws;
  size_t off = 0;
  auto carve = [&](size_t bytes) -> char* {
    char* p = base + off;
    off += (bytes + 255) & ~(size_t)255;
    return p;
  };
  __bf16* actbf = (__bf16*)carve((size_t)GNN_N * 2048 * 2);  // layer input acts (bf16)
  __bf16* hbf   = (__bf16*)carve((size_t)GNN_N * 2048 * 2);  // layer GEMM output (bf16)
  float*  acc   = (float*) carve((size_t)GNN_N * 2048 * 4);  // attention accumulation (f32)
  float*  ssrc  = (float*) carve((size_t)GNN_N * GNN_H * 4);
  float*  sdst  = (float*) carve((size_t)GNN_N * GNN_H * 4);
  unsigned* mkey = (unsigned*)carve((size_t)GNN_N * GNN_H * 4);
  float*  denom = (float*) carve((size_t)GNN_N * GNN_H * 4);
  float*  ebuf  = (float*) carve((size_t)GNN_EDG * GNN_H * 4);
  __bf16* w2bf  = (__bf16*)carve((size_t)1024 * 512 * 2);
  __bf16* w3bf  = (__bf16*)carve((size_t)2048 * 1024 * 2);
  __bf16* w4bf  = (__bf16*)carve((size_t)1024 * 2048 * 2);
  __bf16* fc2bf = (__bf16*)carve((size_t)128 * 256 * 2);
  float*  agg   = (float*) carve(1024 * 4);
  float*  znode = (float*) carve(256 * 4);
  __bf16* z1bf  = hbf;    // [E,256] bf16 reuses hbf region
  __bf16* z2bf  = actbf;  // [E,128] bf16 reuses actbf region

  // Weight conversion to bf16 (deterministic each call)
  auto cvt = [&](const float* s, __bf16* d, size_t n) {
    gnn_f32_to_bf16<<<(unsigned)((n + 255) / 256), 256, 0, stream>>>(s, d, n);
  };
  cvt(W2, w2bf, (size_t)1024 * 512);
  cvt(W3, w3bf, (size_t)2048 * 1024);
  cvt(W4, w4bf, (size_t)1024 * 2048);
  cvt(fc2w, fc2bf, (size_t)128 * 256);

  const int EHG = GNN_EDG * GNN_H;
  auto run_attention = [&](int M, int cshift, const float* as, const float* ad,
                           const float* bias, float* aggp) {
    int C = 1 << cshift;
    size_t tot = (size_t)GNN_N * M;
    gnn_scores<<<GNN_N, 128, 0, stream>>>(hbf, as, ad, ssrc, sdst, C, M);
    gnn_zero_u32<<<4096, 256, 0, stream>>>((unsigned*)acc, tot);
    gnn_zero_u32<<<64, 256, 0, stream>>>(mkey, (size_t)GNN_N * GNN_H);
    gnn_zero_u32<<<64, 256, 0, stream>>>((unsigned*)denom, (size_t)GNN_N * GNN_H);
    gnn_edge_logits<<<(EHG + 255) / 256, 256, 0, stream>>>(ei, ssrc, sdst, ebuf, mkey);
    gnn_edge_exp<<<(EHG + 255) / 256, 256, 0, stream>>>(ei, ebuf, mkey, denom);
    gnn_edge_scatter<<<GNN_EDG, 256, 0, stream>>>(ei, hbf, ebuf, denom, acc, M, cshift);
    gnn_finalize<<<(unsigned)((tot + 255) / 256), 256, 0, stream>>>(acc, bias, actbf, aggp,
                                                                    M - 1, tot);
  };
  auto gemm = [&](const __bf16* A, const __bf16* B, __bf16* Cm, int nr, int K, int M) {
    int blocks = (nr >> 7) * (M >> 6);   // 128 rows x 64 cols of C per block
    gnn_gemm_bf16nt<<<blocks, 256, 0, stream>>>(A, B, Cm, nr, K, M);
  };

  // Layer 1 (K=1 outer product) ------------------------------------------------
  gnn_l1<<<(GNN_N * 512 + 255) / 256, 256, 0, stream>>>(x, W1, hbf);
  run_attention(512, 7, as1, ad1, b1, nullptr);
  // Layer 2 --------------------------------------------------------------------
  gemm(actbf, w2bf, hbf, GNN_N, 512, 1024);
  run_attention(1024, 8, as2, ad2, b2, nullptr);
  // Layer 3 --------------------------------------------------------------------
  gemm(actbf, w3bf, hbf, GNN_N, 1024, 2048);
  run_attention(2048, 9, as3, ad3, b3, nullptr);
  // Layer 4 (accumulate channel sums for the mean) -----------------------------
  gemm(actbf, w4bf, hbf, GNN_N, 2048, 1024);
  gnn_zero_u32<<<4, 256, 0, stream>>>((unsigned*)agg, 1024);
  run_attention(1024, 8, as4, ad4, b4, agg);

  // MLP head -------------------------------------------------------------------
  gnn_znode<<<1, 256, 0, stream>>>(agg, fc1w, znode);
  gnn_fc1<<<GNN_E, 256, 0, stream>>>(eattr, fc1w, fc1b, znode, z1bf);
  gemm(z1bf, fc2bf, z2bf, GNN_E, 256, 128);
  gnn_fc3<<<GNN_E, 128, 0, stream>>>(z2bf, fc2b, fc3w, fc3b, (float*)d_out);
}
